// MultiHeadAttention_35287451304631
// MI455X (gfx1250) — matrix operand order repaired and checked by reading, not compiled, not run on hardware
//
#include <hip/hip_runtime.h>

typedef __attribute__((ext_vector_type(16))) _Float16 v16h;
typedef __attribute__((ext_vector_type(8)))  _Float16 v8h;
typedef __attribute__((ext_vector_type(4)))  _Float16 v4h;
typedef __attribute__((ext_vector_type(8)))  float    v8f;
typedef __attribute__((ext_vector_type(4)))  float    v4f;

#define WMMA_F16(a, b, c) \
  __builtin_amdgcn_wmma_f32_16x16x32_f16(false, (a), false, (b), (short)0, (c), false, false)

#define CAT16(lo, hi) \
  __builtin_shufflevector((lo), (hi), 0, 1, 2, 3, 4, 5, 6, 7, 8, 9, 10, 11, 12, 13, 14, 15)

// ---------------------------------------------------------------------------
// CDNA5 async global->LDS copy (ASYNCcnt path, bypasses VGPRs).
// dst: 32-bit LDS byte address (low 32 bits of generic shared pointer)
// src: 64-bit global address
// ---------------------------------------------------------------------------
__device__ __forceinline__ void async_copy_b128(unsigned lds_addr, const void* gptr) {
  asm volatile("global_load_async_to_lds_b128 %0, %1, off"
               :: "v"(lds_addr), "v"((unsigned long long)(size_t)gptr)
               : "memory");
}
__device__ __forceinline__ void wait_async0() {
  asm volatile("s_wait_asynccnt 0" ::: "memory");
}

// ---------------------------------------------------------------------------
// Fragment loaders (wave32 CDNA5 layouts), all 16-byte contiguous per lane.
// ---------------------------------------------------------------------------
__device__ __forceinline__ v16h load_a_frag(const _Float16* base, int ld, int lane) {
  const _Float16* p = base + (lane & 15) * ld + (lane >> 4) * 8;
  v8h lo = *(const v8h*)(p);
  v8h hi = *(const v8h*)(p + 16);
  return CAT16(lo, hi);
}

// B 32x16 from k-contiguous storage: element (k,n) = base[n*ld + k]
__device__ __forceinline__ v16h load_bk_frag(const _Float16* base, int ld, int lane) {
  const _Float16* p = base + (lane & 15) * ld + (lane >> 4) * 8;
  v8h lo = *(const v8h*)(p);
  v8h hi = *(const v8h*)(p + 16);
  return CAT16(lo, hi);
}

// ---------------------------------------------------------------------------
// Kernel 1: per-head QKV projection (software-pipelined, double-buffered LDS).
//   q,k -> [B,H,S,D] f16 ; v -> [B,H,D,S] f16 (pre-transposed for PV matmul)
// ---------------------------------------------------------------------------
__global__ __launch_bounds__(256)
void qkv_kernel(const float* __restrict__ x,
                const float* __restrict__ Wq, const float* __restrict__ bq,
                const float* __restrict__ Wk, const float* __restrict__ bk,
                const float* __restrict__ Wv, const float* __restrict__ bv,
                _Float16* __restrict__ qo, _Float16* __restrict__ ko,
                _Float16* __restrict__ vo) {
  const int m0 = blockIdx.x * 64;
  const int h  = blockIdx.y;
  const int z  = blockIdx.z;                  // 0=q, 1=k, 2=v
  const float* W    = (z == 0) ? Wq : (z == 1) ? Wk : Wv;
  const float* bias = (z == 0) ? bq : (z == 1) ? bk : bv;
  _Float16*    out  = (z == 0) ? qo : (z == 1) ? ko : vo;

  __shared__ _Float16 Xs[2][64][40];    // 64 rows x 32 k (+pad)
  __shared__ _Float16 WsT[2][64][40];   // 64 n x 32 k (+pad), k-contiguous

  const int tid  = threadIdx.x;
  const int lane = tid & 31;
  const int wave = tid >> 5;
  const int wm   = (wave >> 1) * 16;
  const int wn   = (wave & 1) * 32;
  const int wln  = tid & 63;
  const int wkg  = tid >> 6;

  v4h xr[2];
  v8h wv;

  auto load_X = [&](int k0) {
#pragma unroll
    for (int i = 0; i < 2; ++i) {
      int idx = tid + i * 256;
      int r = idx >> 3, c4 = (idx & 7) * 4;
      v4f xv = *(const v4f*)(x + (size_t)(m0 + r) * 1024 + k0 + c4);
      v4h xh;
      xh[0] = (_Float16)xv[0]; xh[1] = (_Float16)xv[1];
      xh[2] = (_Float16)xv[2]; xh[3] = (_Float16)xv[3];
      xr[i] = xh;
    }
  };
  auto load_W = [&](int k0) {
#pragma unroll
    for (int kk = 0; kk < 8; ++kk)
      wv[kk] = (_Float16)W[((size_t)h * 1024 + k0 + wkg * 8 + kk) * 64 + wln];
  };

  v8f c0 = {};
  v8f c1 = {};

  load_X(0);
  load_W(0);
  for (int kt = 0; kt < 32; ++kt) {
    const int buf = kt & 1;
    // commit staged registers to LDS
#pragma unroll
    for (int i = 0; i < 2; ++i) {
      int idx = tid + i * 256;
      int r = idx >> 3, c4 = (idx & 7) * 4;
      *(v4h*)&Xs[buf][r][c4] = xr[i];
    }
    *(v8h*)&WsT[buf][wln][wkg * 8] = wv;
    __syncthreads();
    if (kt < 31) { load_X((kt + 1) * 32); load_W((kt + 1) * 32); }  // overlap

    v16h aa = load_a_frag(&Xs[buf][wm][0], 40, lane);
    v16h b0 = load_bk_frag(&WsT[buf][wn][0], 40, lane);
    v16h b1 = load_bk_frag(&WsT[buf][wn + 16][0], 40, lane);
    c0 = WMMA_F16(aa, b0, c0);
    c1 = WMMA_F16(aa, b1, c1);
  }

  const int nl    = lane & 15;
  const int mbase = m0 + wm + (lane >> 4) * 8;
  const int bb    = mbase >> 11;
  const int s0    = mbase & 2047;

  if (z == 2) {
    // v: transposed store [B,H,D,S]; lane owns fixed d, 8 consecutive s -> b128
    float bv0 = bias[h * 64 + wn + nl];
    float bv1 = bias[h * 64 + wn + 16 + nl];
    v8h o0, o1;
#pragma unroll
    for (int r = 0; r < 8; ++r) {
      o0[r] = (_Float16)(c0[r] + bv0);
      o1[r] = (_Float16)(c1[r] + bv1);
    }
    size_t dbase = ((size_t)bb * 16 + h) * 64;
    *(v8h*)&out[(dbase + wn + nl) * 2048 + s0]      = o0;
    *(v8h*)&out[(dbase + wn + 16 + nl) * 2048 + s0] = o1;
  } else {
#pragma unroll
    for (int r = 0; r < 8; ++r) {
      size_t base = (((size_t)bb * 16 + h) * 2048 + (s0 + r)) * 64;
      out[base + wn + nl]      = (_Float16)(c0[r] + bias[h * 64 + wn + nl]);
      out[base + wn + 16 + nl] = (_Float16)(c1[r] + bias[h * 64 + wn + 16 + nl]);
    }
  }
}

// ---------------------------------------------------------------------------
// Kernel 2: causal flash attention per (b,h).  Block = 128 thr (4 waves).
// K/V tiles staged with async global->LDS (ASYNCcnt), double-buffered.
// v arrives pre-transposed [B,H,D,S].  Output f16 [B,S,E], e = h*64+d.
// ---------------------------------------------------------------------------
__global__ __launch_bounds__(128)
void attn_kernel(const _Float16* __restrict__ q,
                 const _Float16* __restrict__ k,
                 const _Float16* __restrict__ v,
                 _Float16* __restrict__ o) {
  const int r0 = blockIdx.x * 64;
  const int h  = blockIdx.y;
  const int b  = blockIdx.z;
  const int tid = threadIdx.x, lane = tid & 31, wave = tid >> 5;

  __shared__ _Float16 Ks[2][64][72];    // [key][d]  (d contiguous)
  __shared__ _Float16 VsT[2][64][72];   // [d][key]  (key contiguous)
  __shared__ _Float16 Ps[4][16][64];    // per-wave P (rows x keys)

  const size_t headbase = ((size_t)b * 16 + h) * 2048 * 64;
  const _Float16* kh = k + headbase;   // [S][D]
  const _Float16* vh = v + headbase;   // [D][S]

  auto issue_tile = [&](int t, int buf) {
#pragma unroll
    for (int i = 0; i < 4; ++i) {
      int idx = tid + i * 128;                 // 512 x b128 chunks per tile pair
      int r = idx >> 3, c8 = (idx & 7) * 8;
      async_copy_b128((unsigned)(size_t)&Ks[buf][r][c8],
                      kh + (size_t)(t * 64 + r) * 64 + c8);
      async_copy_b128((unsigned)(size_t)&VsT[buf][r][c8],
                      vh + (size_t)r * 2048 + t * 64 + c8);
    }
  };

  issue_tile(0, 0);   // loads in flight while we build Q fragments

  // Q fragments in A layout, four b128 global loads per wave-row
  v16h qa0, qa1;
  {
    const int row = r0 + wave * 16 + (lane & 15);
    const _Float16* qp = q + headbase + (size_t)row * 64 + (lane >> 4) * 8;
    v8h q0 = *(const v8h*)(qp);
    v8h q1 = *(const v8h*)(qp + 16);
    v8h q2 = *(const v8h*)(qp + 32);
    v8h q3 = *(const v8h*)(qp + 48);
    qa0 = CAT16(q0, q1);
    qa1 = CAT16(q2, q3);
  }

  v8f acc[4] = {{}, {}, {}, {}};
  float mrow[8], lrow[8];
#pragma unroll
  for (int r = 0; r < 8; ++r) { mrow[r] = -1e30f; lrow[r] = 0.f; }

  const float scale = 0.125f;
  const int   tmax  = r0 >> 6;

  for (int t = 0; t <= tmax; ++t) {
    const int buf = t & 1;
    wait_async0();        // my tile-t async copies done
    __syncthreads();      // everyone's done -> LDS tile complete
    if (t < tmax) issue_tile(t + 1, buf ^ 1);   // overlap next tile with compute

    // S = Q @ K^T
    v8f sfr[4];
#pragma unroll
    for (int n = 0; n < 4; ++n) {
      v8f cf = {};
      v16h b0 = load_bk_frag(&Ks[buf][n * 16][0], 72, lane);
      v16h b1 = load_bk_frag(&Ks[buf][n * 16][32], 72, lane);
      cf = WMMA_F16(qa0, b0, cf);
      cf = WMMA_F16(qa1, b1, cf);
      sfr[n] = cf;
    }

    // scale + causal mask
    const int myrow0 = r0 + wave * 16 + (lane >> 4) * 8;
#pragma unroll
    for (int n = 0; n < 4; ++n) {
      int key = t * 64 + n * 16 + (lane & 15);
#pragma unroll
      for (int r = 0; r < 8; ++r) {
        float sv = sfr[n][r] * scale;
        sfr[n][r] = (key <= myrow0 + r) ? sv : -1e30f;
      }
    }

    // online softmax (row reductions within 16-lane halves)
    float newm[8], alpha[8];
#pragma unroll
    for (int r = 0; r < 8; ++r) {
      float mx = fmaxf(fmaxf(sfr[0][r], sfr[1][r]), fmaxf(sfr[2][r], sfr[3][r]));
#pragma unroll
      for (int off = 1; off < 16; off <<= 1)
        mx = fmaxf(mx, __shfl_xor(mx, off, 32));
      newm[r]  = fmaxf(mrow[r], mx);
      alpha[r] = __expf(mrow[r] - newm[r]);
    }
#pragma unroll
    for (int n = 0; n < 4; ++n)
#pragma unroll
      for (int r = 0; r < 8; ++r)
        sfr[n][r] = __expf(sfr[n][r] - newm[r]);
#pragma unroll
    for (int r = 0; r < 8; ++r) {
      float sm = sfr[0][r] + sfr[1][r] + sfr[2][r] + sfr[3][r];
#pragma unroll
      for (int off = 1; off < 16; off <<= 1)
        sm += __shfl_xor(sm, off, 32);
      lrow[r] = lrow[r] * alpha[r] + sm;
      mrow[r] = newm[r];
    }

    // stage P (f16): wave-private LDS region, per-wave DS ops are in-order,
    // so no block barrier is needed for this C-layout -> A-layout transpose.
    {
      const int rl = (lane >> 4) * 8;
#pragma unroll
      for (int n = 0; n < 4; ++n)
#pragma unroll
        for (int r = 0; r < 8; ++r)
          Ps[wave][rl + r][n * 16 + (lane & 15)] = (_Float16)sfr[n][r];
    }

    // acc = acc*alpha + P @ V
#pragma unroll
    for (int n = 0; n < 4; ++n)
#pragma unroll
      for (int r = 0; r < 8; ++r)
        acc[n][r] *= alpha[r];

    v16h pa0 = load_a_frag(&Ps[wave][0][0], 64, lane);
    v16h pa1 = load_a_frag(&Ps[wave][0][32], 64, lane);
#pragma unroll
    for (int n = 0; n < 4; ++n) {
      v16h b0 = load_bk_frag(&VsT[buf][n * 16][0], 72, lane);
      v16h b1 = load_bk_frag(&VsT[buf][n * 16][32], 72, lane);
      acc[n] = WMMA_F16(pa0, b0, acc[n]);
      acc[n] = WMMA_F16(pa1, b1, acc[n]);
    }
  }

  // normalize + store f16 to [B,S,E]
  const int nl  = lane & 15;
  const int rl0 = r0 + wave * 16 + (lane >> 4) * 8;
#pragma unroll
  for (int r = 0; r < 8; ++r) {
    float inv = 1.0f / lrow[r];
    size_t base = ((size_t)b * 2048 + (rl0 + r)) * 1024 + h * 64;
#pragma unroll
    for (int n = 0; n < 4; ++n)
      o[base + n * 16 + nl] = (_Float16)(acc[n][r] * inv);
  }
}

// ---------------------------------------------------------------------------
// Kernel 3: output projection (async A-tile staging, double-buffered).
//   out[8192,1024] (f32) = attn(f16) @ Wp + bp
// ---------------------------------------------------------------------------
__global__ __launch_bounds__(256)
void proj_kernel(const _Float16* __restrict__ a,   // [8192,1024] f16
                 const float* __restrict__ Wp,     // [1024,1024]
                 const float* __restrict__ bp,
                 float* __restrict__ out) {
  const int m0 = blockIdx.x * 64;
  const int n0 = blockIdx.y * 64;

  __shared__ _Float16 As[2][64][40];
  __shared__ _Float16 BsT[2][64][40];

  const int tid = threadIdx.x, lane = tid & 31, wave = tid >> 5;
  const int wm = (wave >> 1) * 16, wn = (wave & 1) * 32;
  const int wln = tid & 63, wkg = tid >> 6;

  auto issue_A = [&](int k0, int buf) {
    int r = tid >> 2, c8 = (tid & 3) * 8;
    async_copy_b128((unsigned)(size_t)&As[buf][r][c8],
                    a + (size_t)(m0 + r) * 1024 + k0 + c8);
  };
  v8h wv;
  auto load_W = [&](int k0) {
#pragma unroll
    for (int kk = 0; kk < 8; ++kk)
      wv[kk] = (_Float16)Wp[(size_t)(k0 + wkg * 8 + kk) * 1024 + n0 + wln];
  };

  v8f c0 = {}, c1 = {};

  issue_A(0, 0);
  load_W(0);
  for (int kt = 0; kt < 32; ++kt) {
    const int buf = kt & 1;
    *(v8h*)&BsT[buf][wln][wkg * 8] = wv;
    wait_async0();
    __syncthreads();
    if (kt < 31) { issue_A((kt + 1) * 32, buf ^ 1); load_W((kt + 1) * 32); }

    v16h af = load_a_frag(&As[buf][wm][0], 40, lane);
    v16h b0 = load_bk_frag(&BsT[buf][wn][0], 40, lane);
    v16h b1 = load_bk_frag(&BsT[buf][wn + 16][0], 40, lane);
    c0 = WMMA_F16(af, b0, c0);
    c1 = WMMA_F16(af, b1, c1);
  }

  const int nl = lane & 15;
  const int m  = m0 + wm + (lane >> 4) * 8;
#pragma unroll
  for (int r = 0; r < 8; ++r) {
    out[(size_t)(m + r) * 1024 + n0 + wn + nl]      = c0[r] + bp[n0 + wn + nl];
    out[(size_t)(m + r) * 1024 + n0 + wn + 16 + nl] = c1[r] + bp[n0 + wn + 16 + nl];
  }
}

// ---------------------------------------------------------------------------
// Launch: qkv -> flash attention -> output projection
// ---------------------------------------------------------------------------
extern "C" void kernel_launch(void* const* d_in, const int* in_sizes, int n_in,
                              void* d_out, int out_size, void* d_ws, size_t ws_size,
                              hipStream_t stream) {
  const float* x  = (const float*)d_in[0];
  const float* Wq = (const float*)d_in[1];
  const float* bq = (const float*)d_in[2];
  const float* Wk = (const float*)d_in[3];
  const float* bk = (const float*)d_in[4];
  const float* Wv = (const float*)d_in[5];
  const float* bv = (const float*)d_in[6];
  const float* Wp = (const float*)d_in[7];
  const float* bp = (const float*)d_in[8];
  float* out = (float*)d_out;

  const size_t QKV = (size_t)4 * 16 * 2048 * 64;  // 8388608 elems per tensor
  _Float16* qw = (_Float16*)d_ws;                 // [B,H,S,D]
  _Float16* kw = qw + QKV;                        // [B,H,S,D]
  _Float16* vw = kw + QKV;                        // [B,H,D,S] (transposed)
  _Float16* aw = vw + QKV;                        // [B,S,E]

  qkv_kernel<<<dim3(128, 16, 3), 256, 0, stream>>>(x, Wq, bq, Wk, bk, Wv, bv,
                                                   qw, kw, vw);
  attn_kernel<<<dim3(32, 16, 4), 128, 0, stream>>>(qw, kw, vw, aw);
  proj_kernel<<<dim3(128, 16), 256, 0, stream>>>(aw, Wp, bp, out);
}
